// iAFF_2508260900996
// MI455X (gfx1250) — matrix operand order repaired and checked by reading, not compiled, not run on hardware
//
#include <hip/hip_runtime.h>

// Problem constants (match reference)
#define BT   16
#define C    64
#define MID  32
#define H    128
#define W    128
#define HW   (H*W)
#define KSZ  16
#define ST   8
#define NH   15
#define NW   15
#define L    225        // patches
#define LP   240        // patches padded to N-tile multiple
#define QP   256        // K (patch) dim padded to 32-multiple
#define K1   576        // conv3x3 im2col depth = 9 * 64

typedef __attribute__((ext_vector_type(16))) _Float16 v16h;
typedef __attribute__((ext_vector_type(8)))  _Float16 v8h;
typedef __attribute__((ext_vector_type(8)))  float    v8f;

union Frag { v16h v; v8h h[2]; };

__device__ __forceinline__ v8f wmma16(v16h a, v16h b, v8f c) {
  // v_wmma_f32_16x16x32_f16  (8-arg form: neg_a, A, neg_b, B, c_mod, C, reuse_a, reuse_b)
  return __builtin_amdgcn_wmma_f32_16x16x32_f16(false, a, false, b, (short)0, c, false, false);
}
__device__ __forceinline__ float lrelu(float x) { return x > 0.f ? x : 0.2f * x; }

// ---------------------------------------------------------------------------
// CDNA5 async global->LDS copy (no destination VGPRs, tracked by ASYNCcnt).
// Flat pointers to __shared__ carry the LDS byte address in their low 32 bits
// (ISA 10.2 aperture rule), so a truncating cast gives the VDST operand.
// ---------------------------------------------------------------------------
__device__ __forceinline__ void async_b128(void* lds_dst, const void* g_src) {
  uint32_t l = (uint32_t)(size_t)lds_dst;
  asm volatile("global_load_async_to_lds_b128 %0, %1, off"
               :: "v"(l), "v"(g_src) : "memory");
}
__device__ __forceinline__ void wait_async0() {
  asm volatile("s_wait_asynccnt 0x0" ::: "memory");
}

// ---------------------------------------------------------------------------
// Prep: f16 conversions + transposed/padded layouts so WMMA fragments are
// contiguous 16/32-byte LDS/global loads.
//   w1t  [MID][K1]   k = (ky*3+kx)*64 + cin   (B^T for conv3x3 GEMM)
//   w2t  [C][MID], wlt [C][C], spe16 [BT][C][C]
//   spaB [BT][LP][QP]  spaB[b][p][q] = spa[b,p,q] (zero padded)
// ---------------------------------------------------------------------------
__global__ void k_prep(const float* __restrict__ W1, const float* __restrict__ W2,
                       const float* __restrict__ Wl, const float* __restrict__ spe,
                       const float* __restrict__ spa,
                       _Float16* w1t, _Float16* w2t, _Float16* wlt,
                       _Float16* spe16, _Float16* spaB) {
  int tid  = blockIdx.x * blockDim.x + threadIdx.x;
  int nthr = gridDim.x * blockDim.x;
  for (int i = tid; i < MID*K1; i += nthr) {
    int co = i / K1, k = i % K1;
    int tap = k >> 6, ci = k & 63;
    w1t[i] = (_Float16)W1[((co*C + ci)*3 + tap/3)*3 + tap%3];
  }
  for (int i = tid; i < C*MID;  i += nthr) w2t[i]  = (_Float16)W2[i];
  for (int i = tid; i < C*C;    i += nthr) wlt[i]  = (_Float16)Wl[i];
  for (int i = tid; i < BT*C*C; i += nthr) spe16[i] = (_Float16)spe[i];
  for (int i = tid; i < BT*LP*QP; i += nthr) {
    int b = i / (LP*QP), r = i % (LP*QP);
    int p = r / QP, q = r % QP;
    spaB[i] = (p < L && q < L) ? (_Float16)spa[(b*L + p)*L + q] : (_Float16)0.f;
  }
}

// ---------------------------------------------------------------------------
// BasicBlock: reflect-pad conv3x3 (64->32, im2col WMMA GEMM, K=576) -> lrelu
//             -> conv1x1 (32->64, single 16x16x32 WMMA step) -> lrelu
// One WG per (batch, 16x16 pixel tile). 8 waves, 2 M-tiles each.
// Weights staged via async global->LDS; halo gather (reflect pad + f16
// convert) overlaps with the in-flight async copies.
// ---------------------------------------------------------------------------
__global__ void k_basicblock(const float* __restrict__ fea,
                             const _Float16* __restrict__ w1t_g,
                             const _Float16* __restrict__ w2t_g,
                             float* __restrict__ fout, _Float16* __restrict__ f16out) {
  extern __shared__ __align__(32) char smem[];
  _Float16* in16  = (_Float16*)smem;                           // [324][64]  41472 B
  _Float16* mid16 = (_Float16*)(smem + 41472);                 // [256][32]  16384 B
  _Float16* w1t   = (_Float16*)(smem + 41472 + 16384);         // 36864 B
  _Float16* w2t   = (_Float16*)(smem + 41472 + 16384 + 36864); // 4096 B

  const int b  = blockIdx.y;
  const int ty = blockIdx.x >> 3, tx = blockIdx.x & 7;
  const int tid  = threadIdx.x;
  const int lane = tid & 31, wv = tid >> 5;
  const int lh = lane >> 4, lm = lane & 15;

  // async weight staging: 36864/16 = 2304 and 4096/16 = 256 b128 transfers
  for (int i = tid; i < 2304; i += 256) async_b128(w1t + i*8, w1t_g + i*8);
  if (tid < 256) async_b128(w2t + tid*8, w2t_g + tid*8);

  // 18x18 halo with reflect padding, channel-innermost for contiguous A frags
  for (int c = 0; c < C; ++c) {
    const float* plane = fea + (b*C + c)*HW;
    for (int p = tid; p < 324; p += 256) {
      int iy = p / 18, ix = p % 18;
      int gy = ty*16 - 1 + iy; gy = gy < 0 ? -gy : (gy > 127 ? 254 - gy : gy);
      int gx = tx*16 - 1 + ix; gx = gx < 0 ? -gx : (gx > 127 ? 254 - gx : gx);
      in16[p*64 + c] = (_Float16)plane[gy*W + gx];
    }
  }
  wait_async0();
  __syncthreads();

  // conv3x3: M=256 pixels (16 tiles), N=32 (2 tiles), K=576 (18 steps)
  for (int mi = 0; mi < 2; ++mi) {
    int mt = wv*2 + mi;
    int py = mt, px = lm;                              // A-row pixel (row = lm)
    for (int nt = 0; nt < 2; ++nt) {
      v8f acc = {};
      for (int ks = 0; ks < 18; ++ks) {
        Frag a;
        int k0 = ks*32 + lh*16;                        // lane's first 8-half chunk
        int t0 = k0 >> 6, c0 = k0 & 63;
        a.h[0] = *(const v8h*)&in16[((py + t0/3)*18 + (px + t0%3))*64 + c0];
        int k1 = k0 + 8;                               // lane's second chunk
        int t1 = k1 >> 6, c1 = k1 & 63;
        a.h[1] = *(const v8h*)&in16[((py + t1/3)*18 + (px + t1%3))*64 + c1];
        v16h bm = *(const v16h*)&w1t[(nt*16 + lm)*K1 + ks*32 + lh*16];
        acc = wmma16(a.v, bm, acc);
      }
      for (int r = 0; r < 8; ++r) {
        int m = mt*16 + r + lh*8;                      // D row -> pixel
        int n = nt*16 + lm;                            // D col -> mid channel
        mid16[m*32 + n] = (_Float16)lrelu(acc[r]);
      }
    }
  }
  __syncthreads();

  // conv1x1: M=256, N=64 (4 tiles), K=32 (exactly one WMMA step)
  for (int mi = 0; mi < 2; ++mi) {
    int mt = wv*2 + mi;
    for (int nt = 0; nt < 4; ++nt) {
      v8f acc = {};
      Frag a;
      a.h[0] = *(const v8h*)&mid16[(mt*16 + lm)*32 + lh*16];
      a.h[1] = *(const v8h*)&mid16[(mt*16 + lm)*32 + lh*16 + 8];
      v16h bm = *(const v16h*)&w2t[(nt*16 + lm)*32 + lh*16];
      acc = wmma16(a.v, bm, acc);
      for (int r = 0; r < 8; ++r) {
        int m  = mt*16 + r + lh*8;
        int co = nt*16 + lm;
        int gy = ty*16 + (m >> 4), gx = tx*16 + (m & 15);
        float v = lrelu(acc[r]);
        fout[((b*C + co)*H + gy)*W + gx] = v;          // NCHW f32 (residual path)
        f16out[(b*HW + gy*W + gx)*C + co] = (_Float16)v; // NHWC f16 (GEMM B path)
      }
    }
  }
}

// ---------------------------------------------------------------------------
// Spatial graph GEMM + fused fold.  One WG per (batch, channel) plane.
// out_spa[(ky,kx), p] = sum_q cols[(ky,kx), q] * spa[p, q]  (M=256,N=225,K=225)
// never hits HBM: overlap-add folded into a 64 KB LDS plane via ds_add_f32.
// Two half-M passes keep LDS at 128 KB (2 WGs / WGP).
// ---------------------------------------------------------------------------
__global__ void k_spatial(const float* __restrict__ f32g,
                          const _Float16* __restrict__ spaB,
                          float* __restrict__ xout) {
  extern __shared__ __align__(32) char smem[];
  _Float16* colsA = (_Float16*)smem;            // [128][QP]  65536 B
  float*    xacc  = (float*)(smem + 65536);     // [HW]       65536 B

  const int b = blockIdx.y, c = blockIdx.x;
  const int tid  = threadIdx.x;
  const int lane = tid & 31, wv = tid >> 5;
  const int lh = lane >> 4, lm = lane & 15;
  const float*    fplane = f32g + (b*C + c)*HW;
  const _Float16* spab   = spaB + b*LP*QP;

  // warm the L2-resident 120 KB graph panel (one cacheline per iteration)
  for (int i = tid; i < 480; i += 256)
    __builtin_prefetch((const char*)spab + i*256, 0, 1);

  for (int i = tid; i < HW; i += 256) xacc[i] = 0.f;
  __syncthreads();

  for (int pass = 0; pass < 2; ++pass) {
    // stage im2col rows m = pass*128 .. +127 (zero-padded q >= 225)
    for (int i = tid; i < 128*QP; i += 256) {
      int ml = i >> 8, q = i & 255;
      int m  = pass*128 + ml;
      _Float16 v = (_Float16)0.f;
      if (q < L) {
        int qh = q / NW, qw = q % NW;
        v = (_Float16)fplane[(qh*ST + (m >> 4))*W + qw*ST + (m & 15)];
      }
      colsA[ml*QP + q] = v;
    }
    __syncthreads();

    int mtg   = pass*8 + wv;        // global M-tile (one per wave per pass)
    int mlrow = wv*16 + lm;         // local colsA row for A fragment
    for (int nt = 0; nt < 15; ++nt) {
      v8f acc = {};
      for (int ks = 0; ks < 8; ++ks) {
        Frag a;
        a.h[0] = *(const v8h*)&colsA[mlrow*QP + ks*32 + lh*16];
        a.h[1] = *(const v8h*)&colsA[mlrow*QP + ks*32 + lh*16 + 8];
        v16h bm = *(const v16h*)&spab[(nt*16 + lm)*QP + ks*32 + lh*16]; // L2-resident
        acc = wmma16(a.v, bm, acc);
      }
      int p = nt*16 + lm;           // D column -> patch index
      if (p < L) {
        int ph = p / NW, pw = p % NW;
        for (int r = 0; r < 8; ++r) {
          int m = mtg*16 + r + lh*8;                 // D row -> (ky,kx)
          atomicAdd(&xacc[(ph*ST + (m >> 4))*W + pw*ST + (m & 15)], acc[r]);
        }
      }
    }
    __syncthreads();
  }

  float* xp = xout + (b*C + c)*HW;
  for (int i = tid; i < HW; i += 256) xp[i] = xacc[i];
}

// ---------------------------------------------------------------------------
// Spectral graph GEMM: residual = spe[b] (64x64) @ f[b] (64 x 16384).
// One WG per (batch, 256-pixel block); B frags straight from NHWC f16 f.
// ---------------------------------------------------------------------------
__global__ void k_spectral(const _Float16* __restrict__ spe16,
                           const _Float16* __restrict__ f16g,
                           float* __restrict__ resout) {
  extern __shared__ __align__(32) char smem[];
  _Float16* spl = (_Float16*)smem;              // [64][64]  8192 B

  const int b = blockIdx.y, blk = blockIdx.x;
  const int tid  = threadIdx.x;
  const int lane = tid & 31, wv = tid >> 5;
  const int lh = lane >> 4, lm = lane & 15;

  // async stage of the 8 KB spectral graph (512 x b128)
  for (int i = tid; i < 512; i += 256)
    async_b128(spl + i*8, spe16 + b*C*C + i*8);
  wait_async0();
  __syncthreads();

  const _Float16* fb = f16g + (size_t)b * HW * C;
  for (int ni = 0; ni < 2; ++ni) {
    int nt  = wv*2 + ni;
    int pix = blk*256 + nt*16 + lm;                     // B-column pixel
    const _Float16* colp = fb + (size_t)pix * C;
    for (int mt = 0; mt < 4; ++mt) {
      v8f acc = {};
      for (int ks = 0; ks < 2; ++ks) {
        Frag a;
        a.h[0] = *(const v8h*)&spl[(mt*16 + lm)*C + ks*32 + lh*16];
        a.h[1] = *(const v8h*)&spl[(mt*16 + lm)*C + ks*32 + lh*16 + 8];
        v16h bm = *(const v16h*)(colp + ks*32 + lh*16); // 32B global load
        acc = wmma16(a.v, bm, acc);
      }
      for (int r = 0; r < 8; ++r) {
        int cc = mt*16 + r + lh*8;
        resout[(b*C + cc)*HW + blk*256 + nt*16 + lm] = acc[r];
      }
    }
  }
}

// ---------------------------------------------------------------------------
// Gate + combine: xa = 0.5(x+res); xl = lrelu(Wl@xa + bl); wei = sigmoid(xl);
// out = x*wei + res*(1-wei) + f.  1x1 conv via WMMA (M=256 pix, N=64, K=64).
// ---------------------------------------------------------------------------
__global__ void k_final(const float* __restrict__ xg, const float* __restrict__ resg,
                        const float* __restrict__ fg,
                        const _Float16* __restrict__ wlt_g,
                        const float* __restrict__ bl, float* __restrict__ out) {
  extern __shared__ __align__(32) char smem[];
  _Float16* xa16 = (_Float16*)smem;              // [256][64]  32768 B
  _Float16* wlt  = (_Float16*)(smem + 32768);    // 8192 B

  const int b = blockIdx.y, blk = blockIdx.x;
  const int tid  = threadIdx.x;
  const int lane = tid & 31, wv = tid >> 5;
  const int lh = lane >> 4, lm = lane & 15;
  const int base = b*C*HW;
  const int pix0 = blk*256;

  // async stage of the 8 KB gating weights; xa gather overlaps with it
  for (int i = tid; i < 512; i += 256)
    async_b128(wlt + i*8, wlt_g + i*8);
  for (int cc = 0; cc < C; ++cc) {
    int gi = base + cc*HW + pix0 + tid;          // coalesced along pixels
    xa16[tid*64 + cc] = (_Float16)(0.5f*(xg[gi] + resg[gi]));
  }
  wait_async0();
  __syncthreads();

  for (int mi = 0; mi < 2; ++mi) {
    int mt = wv*2 + mi;
    for (int nt = 0; nt < 4; ++nt) {
      v8f acc = {};
      for (int ks = 0; ks < 2; ++ks) {
        Frag a;
        a.h[0] = *(const v8h*)&xa16[(mt*16 + lm)*C + ks*32 + lh*16];
        a.h[1] = *(const v8h*)&xa16[(mt*16 + lm)*C + ks*32 + lh*16 + 8];
        v16h bm = *(const v16h*)&wlt[(nt*16 + lm)*C + ks*32 + lh*16];
        acc = wmma16(a.v, bm, acc);
      }
      int co = nt*16 + lm;
      float bb = bl[co];
      for (int r = 0; r < 8; ++r) {
        int m  = mt*16 + r + lh*8;
        int gi = base + co*HW + pix0 + m;
        float xl  = lrelu(acc[r] + bb);
        float wei = 1.f / (1.f + __expf(-xl));
        out[gi] = xg[gi]*wei + resg[gi]*(1.f - wei) + fg[gi];
      }
    }
  }
}

// ---------------------------------------------------------------------------
extern "C" void kernel_launch(void* const* d_in, const int* in_sizes, int n_in,
                              void* d_out, int out_size, void* d_ws, size_t ws_size,
                              hipStream_t stream) {
  (void)in_sizes; (void)n_in; (void)out_size; (void)ws_size;
  const float* fea = (const float*)d_in[0];
  const float* spa = (const float*)d_in[1];
  const float* spe = (const float*)d_in[2];
  const float* W1  = (const float*)d_in[3];
  const float* W2  = (const float*)d_in[4];
  const float* Wl  = (const float*)d_in[5];
  const float* bl  = (const float*)d_in[6];
  float* out = (float*)d_out;

  char* ws = (char*)d_ws;
  size_t off = 0;
  auto carve = [&](size_t bytes) {
    char* p = ws + off;
    off = (off + bytes + 255) & ~(size_t)255;
    return p;
  };
  float*    f32   = (float*)carve(sizeof(float)    * (size_t)BT*C*HW);   // f (NCHW)
  float*    x32   = (float*)carve(sizeof(float)    * (size_t)BT*C*HW);   // folded x
  float*    r32   = (float*)carve(sizeof(float)    * (size_t)BT*C*HW);   // residual
  _Float16* f16g  = (_Float16*)carve(sizeof(_Float16) * (size_t)BT*HW*C); // f (NHWC f16)
  _Float16* spaB  = (_Float16*)carve(sizeof(_Float16) * (size_t)BT*LP*QP);
  _Float16* w1t   = (_Float16*)carve(sizeof(_Float16) * MID*K1);
  _Float16* w2t   = (_Float16*)carve(sizeof(_Float16) * C*MID);
  _Float16* wlt   = (_Float16*)carve(sizeof(_Float16) * C*C);
  _Float16* spe16 = (_Float16*)carve(sizeof(_Float16) * BT*C*C);

  k_prep<<<dim3(2048), dim3(256), 0, stream>>>(W1, W2, Wl, spe, spa,
                                               w1t, w2t, wlt, spe16, spaB);
  k_basicblock<<<dim3(64, BT), dim3(256), 41472 + 16384 + 36864 + 4096, stream>>>(
      fea, w1t, w2t, f32, f16g);
  k_spatial<<<dim3(C, BT), dim3(256), 65536 + 65536, stream>>>(f32, spaB, x32);
  k_spectral<<<dim3(64, BT), dim3(256), 8192, stream>>>(spe16, f16g, r32);
  k_final<<<dim3(64, BT), dim3(256), 32768 + 8192, stream>>>(x32, r32, f32, wlt, bl, out);
}
